// Attention_16767552324249
// MI455X (gfx1250) — matrix operand order repaired and checked by reading, not compiled, not run on hardware
//
#include <hip/hip_runtime.h>
#include <hip/hip_bf16.h>

// ---------------- problem constants (from reference) ----------------
#define B_  4
#define T_  4096
#define D_  2048
#define H_  128
#define BT_ (B_ * T_)          // 16384 rows

typedef __bf16 bf16;
typedef __attribute__((ext_vector_type(16))) __bf16 v16bf;
typedef __attribute__((ext_vector_type(8)))  float  v8f;
typedef __attribute__((ext_vector_type(4)))  unsigned int u32x4;

union VU {
    v16bf v;
    u32x4 q[2];
    bf16  h[16];
};

__device__ __forceinline__ u32x4 ld128(const bf16* p) {
    return *reinterpret_cast<const u32x4*>(p);
}

// A operand, 16x32 bf16 (M x K), src row-major with leading dim ldm (elems).
// Layout used here (contiguous k, not the ISA table order): lanes 0-15 = row (lane), K = 0..15 contiguous; lanes 16-31 = row(lane-16), K = 16..31
__device__ __forceinline__ v16bf load_A16x32(const bf16* base, int ldm, int lane) {
    const int r  = lane & 15;
    const int c0 = (lane < 16) ? 0 : 16;
    const bf16* p = base + (size_t)r * ldm + c0;
    VU u;
    u.q[0] = ld128(p);        // K = c0 .. c0+7
    u.q[1] = ld128(p + 8);    // K = c0+8 .. c0+15
    return u.v;
}

// B operand, 32x16 bf16 (K x N), loaded from B^T stored row-major [N x K] with leading dim ldn.
// Layout used here (contiguous k, not the ISA table order): lanes 0-15 = col n, K=0..15 contiguous; lanes 16-31 = col n, K=16..31
__device__ __forceinline__ v16bf load_BT16(const bf16* bt, int ldn, int lane) {
    const int n  = lane & 15;
    const int k0 = (lane < 16) ? 0 : 16;
    const bf16* p = bt + (size_t)n * ldn + k0;
    VU u;
    u.q[0] = ld128(p);        // K = k0 .. k0+7
    u.q[1] = ld128(p + 8);    // K = k0+8 .. k0+15
    return u.v;
}

__device__ __forceinline__ v8f wmma_bf16(v16bf a, v16bf b, v8f c) {
    return __builtin_amdgcn_wmma_f32_16x16x32_bf16(
        /*neg_a=*/false, a, /*neg_b=*/false, b,
        /*c_mod=*/(short)0, c, /*reuse_a=*/false, /*reuse_b=*/false);
}

__device__ __forceinline__ v8f zero8() {
    v8f z;
#pragma unroll
    for (int i = 0; i < 8; ++i) z[i] = 0.0f;
    return z;
}

// ---------------- CDNA5 async global->LDS (ASYNCcnt path, inline asm) ----------------
// VGLOBAL async encoding: VDST = LDS byte address (VGPR), VADDR = 64-bit global
// address (VGPR pair), no SADDR ("off"). Tracked by ASYNCcnt.
__device__ __forceinline__ void async_copy16(const bf16* gsrc, bf16* ldst) {
    const unsigned lofs = (unsigned)(size_t)ldst;   // generic shared ptr: low 32 bits = LDS offset
    const unsigned long long ga = (unsigned long long)(size_t)gsrc;
    asm volatile("global_load_async_to_lds_b128 %0, %1, off"
                 :: "v"(lofs), "v"(ga) : "memory");
}

__device__ __forceinline__ void wait_async0() {
#if __has_builtin(__builtin_amdgcn_s_wait_asynccnt)
    __builtin_amdgcn_s_wait_asynccnt(0);
#else
    asm volatile("s_wait_asynccnt 0x0" ::: "memory");
#endif
}

// ---------------- elementwise converts ----------------
__global__ void cvt_f32_bf16(const float* __restrict__ in, bf16* __restrict__ out, size_t n) {
    size_t i = (size_t)blockIdx.x * blockDim.x + threadIdx.x;
    size_t stride = (size_t)gridDim.x * blockDim.x;
    for (; i < n; i += stride) out[i] = (bf16)in[i];
}

// W [D x H] f32 -> W^T [H x D] bf16
__global__ void transpose_w(const float* __restrict__ W, bf16* __restrict__ Wt) {
    int idx = blockIdx.x * blockDim.x + threadIdx.x;
    if (idx < D_ * H_) {
        int d = idx / H_;
        int h = idx % H_;
        Wt[(size_t)h * D_ + d] = (bf16)W[idx];
    }
}

// =======================================================================
// QKV projection: 8 waves/WG, each wave owns a 16-row x tile; the per-step
// weight slice W^T[0:128, kc*32 : kc*32+32] is async-staged in LDS and
// shared by all 8 waves (double buffered).
// =======================================================================
#define WPAD 36   // 32 K elems + 4 pad (72B row stride -> conflict-free 16-lane reads)

__global__ __launch_bounds__(256)
void qkv_proj_kernel(const bf16* __restrict__ xb,
                     const bf16* __restrict__ WtQ,
                     const bf16* __restrict__ WtK,
                     const bf16* __restrict__ WtV,
                     bf16* __restrict__ Qb,   // [BT, H] row-major
                     bf16* __restrict__ Kb,   // [BT, H] row-major
                     bf16* __restrict__ Vt)   // [B, H, T]
{
    __shared__ bf16 smW[2][H_][WPAD];        // 2 * 128 * 36 * 2B = 18 KB

    const int tid   = threadIdx.x;
    const int wave  = tid >> 5;
    const int lane  = tid & 31;
    const int tile  = blockIdx.x * 8 + wave;   // 16-row tile of x
    const int which = blockIdx.y;              // 0=Q, 1=K, 2=V

    const bf16* Wt = (which == 0) ? WtQ : (which == 1) ? WtK : WtV;

    // cooperative async fill of one 128x32 weight slice (8 KB = 512 x 16B segs)
    auto issueW = [&](int kc, int buf) {
#pragma unroll
        for (int s = tid; s < 512; s += 256) {
            const int row = s >> 2;            // 0..127
            const int off = (s & 3) * 8;       // elem offset in row (16B segs)
            async_copy16(Wt + (size_t)row * D_ + kc * 32 + off, &smW[buf][row][off]);
        }
    };

    v8f acc[8];
#pragma unroll
    for (int n = 0; n < 8; ++n) acc[n] = zero8();

    const bf16* xrow = xb + (size_t)tile * 16 * D_;

    issueW(0, 0);
    wait_async0();
    __syncthreads();

    for (int kc = 0; kc < D_ / 32; ++kc) {
        const int cur = kc & 1;
        if (kc + 1 < D_ / 32) issueW(kc + 1, cur ^ 1);

        // A tile from global (per-wave), B tiles from shared LDS slice
        v16bf A = load_A16x32(xrow + kc * 32, D_, lane);
        v16bf Bt[8];
#pragma unroll
        for (int n = 0; n < 8; ++n)
            Bt[n] = load_BT16(&smW[cur][n * 16][0], WPAD, lane);
#pragma unroll
        for (int n = 0; n < 8; ++n)
            acc[n] = wmma_bf16(A, Bt[n], acc[n]);

        wait_async0();
        __syncthreads();
    }

    // C/D layout: N = lane&15 (output h within chunk), M = vgpr (+8 for upper lanes)
    const int nloc = lane & 15;
    const int mb   = (lane < 16) ? 0 : 8;
#pragma unroll
    for (int n = 0; n < 8; ++n) {
#pragma unroll
        for (int r = 0; r < 8; ++r) {
            const int row = tile * 16 + mb + r;       // global row in [BT]
            const int h   = n * 16 + nloc;
            const bf16 val = (bf16)acc[n][r];
            if (which == 0) {
                Qb[(size_t)row * H_ + h] = val;
            } else if (which == 1) {
                Kb[(size_t)row * H_ + h] = val;
            } else {
                const int b = row / T_;
                const int t = row % T_;
                Vt[((size_t)b * H_ + h) * T_ + t] = val;
            }
        }
    }
}

// =======================================================================
// Flash attention: 8 waves/WG = 128 queries; 32-key K and V^T chunks are
// async-staged into double-buffered LDS and shared across the waves.
// =======================================================================
#define KT    32
#define KPADK 136   // 128 h elems + 8 pad (272B rows -> conflict-free)
#define VPAD  36    // 32 key elems + 4 pad

__global__ __launch_bounds__(256)
void flash_attn_kernel(const bf16* __restrict__ Qb,
                       const bf16* __restrict__ Kb,
                       const bf16* __restrict__ Vt,
                       float* __restrict__ out)
{
    __shared__ bf16 smK[2][KT][KPADK];    // 2 * 32 * 136 * 2B = 17.0 KB
    __shared__ bf16 smV[2][H_][VPAD];     // 2 * 128 * 36 * 2B = 18.0 KB

    const int tid  = threadIdx.x;
    const int wave = tid >> 5;
    const int lane = tid & 31;
    const int Q0   = blockIdx.x * 128;
    const int b    = blockIdx.y;
    const int q0   = Q0 + wave * 16;

    const float sc = 0.08838834764831845f;   // 1/sqrt(128)
    const float NEG_INF = -__builtin_inff();

    const bf16* gK  = Kb + (size_t)b * T_ * H_;
    const bf16* gVt = Vt + (size_t)b * H_ * T_;

    // K chunk: 32 consecutive rows of [T,H] = contiguous 8KB; V chunk: 128 rows x 64B strided
    auto issueTile = [&](int kt, int buf) {
        const int k0 = kt * KT;
#pragma unroll
        for (int s = tid; s < 512; s += 256) {         // K: 512 x 16B segments
            const int row = s >> 3;                    // 0..63
            const int off = (s & 7) * 8;               // elem offset (16B segs)
            async_copy16(gK + (size_t)(k0 + row) * H_ + off, &smK[buf][row][off]);
        }
#pragma unroll
        for (int s = tid; s < 512; s += 256) {         // V: 128 rows x 4 segs
            const int row = s >> 2;                    // 0..127 (h)
            const int off = (s & 3) * 8;               // key offset within chunk
            async_copy16(gVt + (size_t)row * T_ + k0 + off, &smV[buf][row][off]);
        }
    };

    // Q^T as B operand, preloaded for all 4 K-steps over H=128 (stays in VGPRs)
    v16bf qB[4];
    const bf16* qbase = Qb + ((size_t)b * T_ + q0) * H_;
#pragma unroll
    for (int kc = 0; kc < 4; ++kc) qB[kc] = load_BT16(qbase + kc * 32, H_, lane);

    v8f o[8];
#pragma unroll
    for (int c = 0; c < 8; ++c) o[c] = zero8();

    float m_run = NEG_INF;
    float l_run = 0.0f;

    const int qidx = q0 + (lane & 15);     // this lane's query (N index)
    const int mb   = (lane < 16) ? 0 : 8;  // M offset in C/D tiles

    const int nt = Q0 / KT + 4;            // key tiles covering [0, Q0+128)

    issueTile(0, 0);
    wait_async0();
    __syncthreads();

    for (int kt = 0; kt < nt; ++kt) {
        const int cur = kt & 1;
        const int k0  = kt * KT;
        if (kt + 1 < nt) issueTile(kt + 1, cur ^ 1);

        if (k0 < q0 + 16) {   // this wave still has unmasked keys in the tile
            // ---- S^T = K . Q^T  (A tiles from LDS; two independent chains) ----
            v16bf Ak[8];
#pragma unroll
            for (int kc = 0; kc < 4; ++kc) {
                Ak[kc]     = load_A16x32(&smK[cur][0][0]  + kc * 32, KPADK, lane);
                Ak[4 + kc] = load_A16x32(&smK[cur][16][0] + kc * 32, KPADK, lane);
            }
            v8f s0 = zero8(), s1 = zero8();
#pragma unroll
            for (int kc = 0; kc < 4; ++kc) {
                s0 = wmma_bf16(Ak[kc],     qB[kc], s0);
                s1 = wmma_bf16(Ak[4 + kc], qB[kc], s1);
            }

            // ---- scale + causal mask + online softmax (keys along M: in-lane) ----
            float p0[8], p1[8];
            float mloc = NEG_INF;
#pragma unroll
            for (int r = 0; r < 8; ++r) {
                const int key0 = k0 + mb + r;
                const int key1 = key0 + 16;
                float v0 = s0[r] * sc;
                float v1 = s1[r] * sc;
                if (key0 > qidx) v0 = NEG_INF;
                if (key1 > qidx) v1 = NEG_INF;
                p0[r] = v0; p1[r] = v1;
                mloc = fmaxf(mloc, fmaxf(v0, v1));
            }
            mloc = fmaxf(mloc, __shfl_xor(mloc, 16, 32));
            const float m_new = fmaxf(m_run, mloc);
            const float alpha = __expf(m_run - m_new);   // 0 on first tile

            float lsum = 0.0f;
#pragma unroll
            for (int r = 0; r < 8; ++r) {
                p0[r] = __expf(p0[r] - m_new);
                p1[r] = __expf(p1[r] - m_new);
                lsum += p0[r] + p1[r];
            }
            lsum += __shfl_xor(lsum, 16, 32);
            l_run = l_run * alpha + lsum;
            m_run = m_new;

#pragma unroll
            for (int c = 0; c < 8; ++c)
#pragma unroll
                for (int r = 0; r < 8; ++r) o[c][r] *= alpha;

            // ---- build P^T in B-operand layout (K=32 keys, N=16 queries) ----
            VU pbu;
#pragma unroll
            for (int r = 0; r < 8; ++r) {
                const float x0 = __shfl_xor(p0[r], 16, 32);
                const float x1 = __shfl_xor(p1[r], 16, 32);
                const float lo = (lane < 16) ? p0[r] : x1;
                const float hi = (lane < 16) ? x0    : p1[r];
                pbu.h[r]     = (bf16)lo;
                pbu.h[8 + r] = (bf16)hi;
            }
            const v16bf pb = pbu.v;

            // ---- out^T += V^T . P^T  (A = V^T tiles from LDS; 8 indep chains) ----
            v16bf Av[8];
#pragma unroll
            for (int c = 0; c < 8; ++c)
                Av[c] = load_A16x32(&smV[cur][c * 16][0], VPAD, lane);
#pragma unroll
            for (int c = 0; c < 8; ++c)
                o[c] = wmma_bf16(Av[c], pb, o[c]);
        }

        wait_async0();      // my next-tile async copies have landed
        __syncthreads();    // everyone done reading cur + everyone's copies visible
    }

    // epilogue: out[b, qidx, h] = o / l   (o layout: N=query=lane&15, M=h=c*16+mb+r)
    const float inv = 1.0f / l_run;
    float* orow = out + ((size_t)b * T_ + qidx) * H_;
#pragma unroll
    for (int c = 0; c < 8; ++c)
#pragma unroll
        for (int r = 0; r < 8; ++r)
            orow[c * 16 + mb + r] = o[c][r] * inv;
}

// ---------------- host launcher ----------------
extern "C" void kernel_launch(void* const* d_in, const int* in_sizes, int n_in,
                              void* d_out, int out_size, void* d_ws, size_t ws_size,
                              hipStream_t stream) {
    const float* x  = (const float*)d_in[0];
    const float* Wq = (const float*)d_in[1];
    const float* Wk = (const float*)d_in[2];
    const float* Wv = (const float*)d_in[3];
    float* out = (float*)d_out;

    // workspace carve-up (bf16 buffers)
    char* ws = (char*)d_ws;
    bf16* xb  = (bf16*)ws;                        ws += (size_t)BT_ * D_ * sizeof(bf16);
    bf16* WtQ = (bf16*)ws;                        ws += (size_t)H_ * D_ * sizeof(bf16);
    bf16* WtK = (bf16*)ws;                        ws += (size_t)H_ * D_ * sizeof(bf16);
    bf16* WtV = (bf16*)ws;                        ws += (size_t)H_ * D_ * sizeof(bf16);
    bf16* Qb  = (bf16*)ws;                        ws += (size_t)BT_ * H_ * sizeof(bf16);
    bf16* Kb  = (bf16*)ws;                        ws += (size_t)BT_ * H_ * sizeof(bf16);
    bf16* Vt  = (bf16*)ws;                        ws += (size_t)BT_ * H_ * sizeof(bf16);

    // 1) convert x to bf16
    {
        size_t n = (size_t)BT_ * D_;
        cvt_f32_bf16<<<4096, 256, 0, stream>>>(x, xb, n);
    }
    // 2) transpose+convert weights
    {
        int n = D_ * H_;
        int blocks = (n + 255) / 256;
        transpose_w<<<blocks, 256, 0, stream>>>(Wq, WtQ);
        transpose_w<<<blocks, 256, 0, stream>>>(Wk, WtK);
        transpose_w<<<blocks, 256, 0, stream>>>(Wv, WtV);
    }
    // 3) QKV projection (WMMA, LDS-staged weights)
    {
        dim3 grid(BT_ / 16 / 8, 3);
        qkv_proj_kernel<<<grid, 256, 0, stream>>>(xb, WtQ, WtK, WtV, Qb, Kb, Vt);
    }
    // 4) flash attention (WMMA, async LDS double buffering)
    {
        dim3 grid(T_ / 128, B_);
        flash_attn_kernel<<<grid, 256, 0, stream>>>(Qb, Kb, Vt, out);
    }
}
